// IPFluxAttnProcessor2_0_57801669869990
// MI455X (gfx1250) — matrix operand order repaired and checked by reading, not compiled, not run on hardware
//
#include <hip/hip_runtime.h>
#include <hip/hip_bf16.h>
#include <stdint.h>

// ---------------------------------------------------------------------------
// CDNA5 (gfx1250) bf16 WMMA implementation of the IP-Flux attention block.
// wave32, v_wmma_f32_16x16x32_bf16, fp32 accumulation.
// GEMM: 128x128 block tile, B streamed through a 3-buffer LDS pipeline by the
// Tensor Data Mover (prefetch distance 2, s_wait_tensorcnt 1 steady state),
// raw split workgroup barriers (no vmem drain), rotating LDS offsets (no
// modulo), sched_barrier to keep DS-load batches ahead of the WMMA chains.
// ---------------------------------------------------------------------------

typedef __attribute__((ext_vector_type(16))) __bf16 v16bf;
typedef __attribute__((ext_vector_type(8)))  __bf16 v8bf;
typedef __attribute__((ext_vector_type(8)))  float  v8f;
typedef __attribute__((ext_vector_type(4)))  unsigned int u32x4;
typedef __attribute__((ext_vector_type(8)))  int i32x8;
typedef __attribute__((ext_vector_type(4)))  int i32x4;

#define HEADS   24
#define HDIM    128
#define DMODEL  3072
#define S_IMG_C 2048
#define S_REF_C 512
#define S_ENC_C 512
#define SQ_C    (S_ENC_C + S_IMG_C)   // 2560 (query rows of main attention)
#define SK_C    (SQ_C + S_REF_C)      // 3072 (key rows of main attention)
#define RANK_C  128

#define BT_ROWS 128                   // GEMM N-tile rows staged per k-step
#define BT_ELEMS (BT_ROWS * 32)       // elements per LDS B buffer

#if defined(__HIP_DEVICE_COMPILE__) && defined(__gfx1250__) && \
    __has_builtin(__builtin_amdgcn_tensor_load_to_lds) &&      \
    __has_builtin(__builtin_amdgcn_s_wait_tensorcnt)
#define USE_TDM 1
#else
#define USE_TDM 0
#endif

// Raw split workgroup barrier: no implicit vmem-drain fence (LDS production
// is synchronized explicitly via s_wait_tensorcnt / data dependencies).
#if defined(__HIP_DEVICE_COMPILE__) &&                  \
    __has_builtin(__builtin_amdgcn_s_barrier_signal) && \
    __has_builtin(__builtin_amdgcn_s_barrier_wait)
#define WG_BARRIER()                         \
  do {                                       \
    __asm__ __volatile__("" ::: "memory");   \
    __builtin_amdgcn_s_barrier_signal(-1);   \
    __builtin_amdgcn_s_barrier_wait(-1);     \
    __asm__ __volatile__("" ::: "memory");   \
  } while (0)
#else
#define WG_BARRIER() __syncthreads()
#endif

// Scheduling fence: keep batched memory issues ahead of the consumer chain.
#if defined(__HIP_DEVICE_COMPILE__) && __has_builtin(__builtin_amdgcn_sched_barrier)
#define SCHED_FENCE() __builtin_amdgcn_sched_barrier(0)
#else
#define SCHED_FENCE() do {} while (0)
#endif

__device__ __forceinline__ v8f wmma_bf16f32(v16bf a, v16bf b, v8f c) {
  // 8-arg form: (neg_a, A, neg_b, B, c_mod, C, reuse_a, reuse_b)
  return __builtin_amdgcn_wmma_f32_16x16x32_bf16(false, a, false, b, (short)0, c,
                                                 false, false);
}

// A-fragment (16x32 bf16) per-lane loader.
// lanes 0-15 : row = base+lane,    K = {c..c+15}
// lanes 16-31: row = base+lane-16, K = {c+16..c+31}
// Caller passes p = rowptr + kchunk + (lane>=16 ? 8 : 0).
__device__ __forceinline__ v16bf load_afrag16(const __bf16* p) {
  const int lhf = ((__lane_id() & 31) >= 16) ? 1 : 0;
  v8bf lo = *(const v8bf*)(p + lhf * 8);
  v8bf hi = *(const v8bf*)(p + lhf * 8 + 8);
  return __builtin_shufflevector(lo, hi, 0, 1, 2, 3, 4, 5, 6, 7,
                                 8, 9, 10, 11, 12, 13, 14, 15);
}

__device__ __forceinline__ float half_group_max(float v) {
#pragma unroll
  for (int m = 1; m < 16; m <<= 1) v = fmaxf(v, __shfl_xor(v, m, 32));
  return v;
}
__device__ __forceinline__ float half_group_sum(float v) {
#pragma unroll
  for (int m = 1; m < 16; m <<= 1) v += __shfl_xor(v, m, 32);
  return v;
}

#if USE_TDM
// Issue a TDM 2D tile load: BT_ROWS rows x 32 bf16 (row stride Kelems) from
// gaddr into LDS at byte offset lds_off, packed row-major BT_ROWSx32.
__device__ __forceinline__ void tdm_load_b_tile(const __bf16* gaddr,
                                                unsigned lds_off, int Kelems) {
  unsigned long long ga = (unsigned long long)(uintptr_t)gaddr;
  u32x4 g0;
  g0[0] = 1u;                                   // count=1 (valid user D#)
  g0[1] = lds_off;                              // lds_addr (bytes)
  g0[2] = (unsigned)ga;                         // global_addr[31:0]
  g0[3] = (unsigned)(ga >> 32) | 0x80000000u;   // global_addr[56:32] | type=2
  i32x8 g1;
  g1[0] = 0x10000;                              // data_size=1 (2 bytes/elem)
  g1[1] = (int)(32u << 16);                     // tensor_dim0 = 32
  g1[2] = (int)((unsigned)BT_ROWS << 16);       // tensor_dim1 = BT_ROWS
  g1[3] = (int)(32u << 16);                     // tile_dim0 = 32
  g1[4] = BT_ROWS;                              // tile_dim1 = BT_ROWS
  g1[5] = Kelems;                               // tensor_dim0_stride = K
  g1[6] = 0;
  g1[7] = 0;
  i32x4 z4 = {};
#if __clang_major__ >= 23
  i32x8 z8 = {};
  __builtin_amdgcn_tensor_load_to_lds(g0, g1, z4, z4, z8, 0);
#else
  __builtin_amdgcn_tensor_load_to_lds(g0, g1, z4, z4, 0);
#endif
}
#endif

// ---------------------------------------------------------------------------
// fp32 -> bf16 elementwise conversion
// ---------------------------------------------------------------------------
__global__ void cvt_f32_to_bf16(const float* __restrict__ src,
                                __bf16* __restrict__ dst, size_t n) {
  size_t i = (size_t)blockIdx.x * blockDim.x + threadIdx.x;
  size_t stride = (size_t)gridDim.x * blockDim.x;
  for (; i < n; i += stride) dst[i] = (__bf16)src[i];
}

// ---------------------------------------------------------------------------
// C[M][N](f32) = A[M][K](bf16) @ B[N][K]^T(bf16) (+ bias) (+ C if accumulate)
// block = 256 threads = 8 waves; block tile 128(M) x 128(N), k-step 32.
// The shared 128x32 B tile is streamed through a 3-buffer LDS pipeline by
// the TDM; each wave runs 8 WMMAs per k-step (16x128 output strip).
// A fragments register double-buffered. M,N multiples of 128; K of 32.
// ---------------------------------------------------------------------------
__global__ void __launch_bounds__(256)
gemm_bf16_wmma(const __bf16* __restrict__ A, const __bf16* __restrict__ B,
               const float* __restrict__ bias, float* __restrict__ C,
               int M, int N, int K, int accumulate) {
  __shared__ __attribute__((aligned(64))) __bf16 Bs[3][BT_ELEMS];
  const int tid = threadIdx.x;
  const int lane = tid & 31;
  const int wave = tid >> 5;
  const int l4 = lane & 15;
  const int hf = (lane >= 16) ? 1 : 0;
  const int m0 = blockIdx.y * 128 + wave * 16;
  const int n0 = blockIdx.x * 128;
  const int nsteps = K / 32;

  const __bf16* Btile = B + (size_t)n0 * K;  // BT_ROWS rows x K
  const __bf16* arow = A + (size_t)(m0 + l4) * K + (hf ? 8 : 0);
  const unsigned lds_base = (unsigned)(uintptr_t)&Bs[0][0];

  // rotating LDS buffer offsets (elements) -- no modulo in the hot loop
  int cur = 0, nxt = BT_ELEMS, spare = 2 * BT_ELEMS;

#if USE_TDM
  // ---- prologue: launch TDM for tiles 0 and 1, preload first A fragment
  if (wave == 0) {
    tdm_load_b_tile(Btile, lds_base, K);
    if (nsteps > 1) tdm_load_b_tile(Btile + 32, lds_base + BT_ELEMS * 2, K);
  }
  v16bf a_cur = load_afrag16(arow);
  if (nsteps > 1) __builtin_amdgcn_s_wait_tensorcnt(1);
  else            __builtin_amdgcn_s_wait_tensorcnt(0);
  WG_BARRIER();
#else
  const int srow = tid >> 1;          // 0..127
  const int scol = (tid & 1) * 16;    // 0,16
  const __bf16* Bstage = Btile + (size_t)srow * K + scol;
  v16bf stg = *(const v16bf*)(Bstage);
  v16bf a_cur = load_afrag16(arow);
  *(v16bf*)&Bs[0][srow * 32 + scol] = stg;
  __syncthreads();
#endif

  v8f acc[8] = {};
  for (int s = 0; s < nsteps; ++s) {
    const int kb = s * 32;
    const bool has_next = (s + 1 < nsteps);

    // ---- issue staging for tile s+2 into the spare buffer
#if USE_TDM
    if (s + 2 < nsteps && wave == 0)
      tdm_load_b_tile(Btile + (size_t)kb + 64,
                      lds_base + (unsigned)spare * 2, K);
#else
    v16bf stg2;
    if (has_next) stg2 = *(const v16bf*)(Bstage + kb + 32);
#endif
    v16bf a_nxt;
    if (has_next) {
      a_nxt = load_afrag16(arow + kb + 32);
      if (kb + 64 < K) __builtin_prefetch(arow + kb + 64, 0, 1);
    }

    // ---- batch the 8 LDS B-fragment loads, then run the WMMA chain
    const __bf16* bp = &Bs[0][0] + cur;
    v16bf bfr[8];
#pragma unroll
    for (int nt = 0; nt < 8; ++nt)
      bfr[nt] = *(const v16bf*)(bp + (nt * 16 + l4) * 32 + (hf ? 16 : 0));
    SCHED_FENCE();
#pragma unroll
    for (int nt = 0; nt < 8; ++nt)
      acc[nt] = wmma_bf16f32(a_cur, bfr[nt], acc[nt]);
    a_cur = a_nxt;

    // ---- rotate buffers; make tile s+1 consumable by everyone
    int t = cur; cur = nxt; nxt = spare; spare = t;
#if USE_TDM
    if (s + 2 < nsteps) __builtin_amdgcn_s_wait_tensorcnt(1);
    else                __builtin_amdgcn_s_wait_tensorcnt(0);
    WG_BARRIER();
#else
    __syncthreads();  // all reads of the target buffer are done
    if (has_next) *(v16bf*)&Bs[0][cur + srow * 32 + scol] = stg2;
    __syncthreads();
#endif
  }

  // ---- epilogue (uniform branches hoisted out of the store loops)
  float bcol[8];
  if (bias) {
#pragma unroll
    for (int nt = 0; nt < 8; ++nt) bcol[nt] = bias[n0 + nt * 16 + l4];
  } else {
#pragma unroll
    for (int nt = 0; nt < 8; ++nt) bcol[nt] = 0.0f;
  }
  if (accumulate) {
#pragma unroll
    for (int nt = 0; nt < 8; ++nt) {
#pragma unroll
      for (int r = 0; r < 8; ++r) {
        size_t idx = (size_t)(m0 + r + hf * 8) * (size_t)N + (n0 + nt * 16 + l4);
        C[idx] += acc[nt][r] + bcol[nt];
      }
    }
  } else {
#pragma unroll
    for (int nt = 0; nt < 8; ++nt) {
#pragma unroll
      for (int r = 0; r < 8; ++r) {
        size_t idx = (size_t)(m0 + r + hf * 8) * (size_t)N + (n0 + nt * 16 + l4);
        C[idx] = acc[nt][r] + bcol[nt];
      }
    }
  }
}

// ---------------------------------------------------------------------------
// Fused per-head post-processing: optional RMS norm (head dim 128), optional
// RoPE, bf16 pack into head-major [h][row][128] (tr=0) or transposed
// [h][d][row] (tr=1) layout; optional scaled second destination.
// grid = (S, HEADS), block = 128
// ---------------------------------------------------------------------------
__global__ void __launch_bounds__(128)
qkv_post(const float* __restrict__ X, const float* __restrict__ normw,
         const float* __restrict__ cosT, const float* __restrict__ sinT,
         int pos0, __bf16* __restrict__ dst1, int cap1, int row0_1, int tr1,
         __bf16* __restrict__ dst2, int cap2, int row0_2, int tr2,
         const float* __restrict__ scale2) {
  const int r = blockIdx.x;
  const int h = blockIdx.y;
  const int d = threadIdx.x;
  __shared__ float sh[HDIM];
  float x = X[(size_t)r * DMODEL + h * HDIM + d];
  if (normw) {
    sh[d] = x * x;
    __syncthreads();
#pragma unroll
    for (int s = 64; s > 0; s >>= 1) {
      if (d < s) sh[d] += sh[d + s];
      __syncthreads();
    }
    float ms = sh[0] * (1.0f / HDIM) + 1e-6f;
    x = x * rsqrtf(ms) * normw[d];
    __syncthreads();
  }
  if (cosT) {
    sh[d] = x;
    __syncthreads();
    float rot = (d & 1) ? sh[d - 1] : -sh[d + 1];
    size_t pr = (size_t)(pos0 + r) * HDIM + d;
    x = x * cosT[pr] + rot * sinT[pr];
  }
  size_t i1 = tr1 ? ((size_t)(h * HDIM + d) * cap1 + (size_t)(row0_1 + r))
                  : ((size_t)h * cap1 * HDIM + (size_t)(row0_1 + r) * HDIM + d);
  dst1[i1] = (__bf16)x;
  if (dst2) {
    float s = scale2 ? scale2[0] : 1.0f;
    size_t i2 = tr2 ? ((size_t)(h * HDIM + d) * cap2 + (size_t)(row0_2 + r))
                    : ((size_t)h * cap2 * HDIM + (size_t)(row0_2 + r) * HDIM + d);
    dst2[i2] = (__bf16)(x * s);
  }
}

// ---------------------------------------------------------------------------
// WMMA flash attention. One wave per 16-row q tile. K stored [h][row][128],
// V stored pre-transposed [h][d][row]. Online softmax with 16-lane shfl
// reductions; P converted C-layout -> A-layout via a 1KB LDS tile.
// Single-wave workgroup: LDS ops are architecturally in-order per wave, so
// no barrier is needed for the P round-trip. sched_barriers keep the batched
// K/V fragment loads issued ahead of their consumer chains.
// Additive key mask: +1 if amask[col]>0 (col<mask_len), +1 for ref tail.
// Output row<split -> dst_lo[row], else dst_hi[row-split] ([.][3072] bf16).
// ---------------------------------------------------------------------------
__global__ void __launch_bounds__(32)
flash_attn_wmma(const __bf16* __restrict__ Q, const __bf16* __restrict__ K,
                const __bf16* __restrict__ Vt, int cap_q, int cap_k, int Sk,
                const float* __restrict__ amask, int mask_len,
                __bf16* __restrict__ dst_lo, __bf16* __restrict__ dst_hi,
                int split) {
  __shared__ __attribute__((aligned(64))) __bf16 Pl[16 * 32];
  const int h = blockIdx.y;
  const int q0 = blockIdx.x * 16;
  const int lane = threadIdx.x;
  const int l4 = lane & 15;
  const int hf = (lane >= 16) ? 1 : 0;

  const __bf16* Qh = Q + (size_t)h * cap_q * HDIM;
  const __bf16* Kh = K + (size_t)h * cap_k * HDIM;
  const __bf16* Vh = Vt + (size_t)h * HDIM * cap_k;

  v16bf aq[4];
  {
    const __bf16* qr = Qh + (size_t)(q0 + l4) * HDIM + (hf ? 8 : 0);
#pragma unroll
    for (int c = 0; c < 4; ++c) aq[c] = load_afrag16(qr + c * 32);
  }

  v8f acc[8] = {};
  float mrow[8], lrow[8];
#pragma unroll
  for (int r = 0; r < 8; ++r) {
    mrow[r] = -3.0e38f;
    lrow[r] = 0.0f;
  }
  const float qk_scale = 0.08838834764831845f;  // 1/sqrt(128)

  for (int kb = 0; kb < Sk; kb += 32) {
    // ---- batch all K fragments, then run the score WMMA chain
    v16bf bk[8];
#pragma unroll
    for (int c = 0; c < 4; ++c) {
      bk[c]     = *(const v16bf*)(Kh + (size_t)(kb + l4) * HDIM + c * 32 + (hf ? 16 : 0));
      bk[4 + c] = *(const v16bf*)(Kh + (size_t)(kb + 16 + l4) * HDIM + c * 32 + (hf ? 16 : 0));
    }
    SCHED_FENCE();
    v8f s0 = {}, s1 = {};
#pragma unroll
    for (int c = 0; c < 4; ++c) {
      s0 = wmma_bf16f32(aq[c], bk[c], s0);
      s1 = wmma_bf16f32(aq[c], bk[4 + c], s1);
    }
    float madd0 = 0.0f, madd1 = 0.0f;
    if (amask) {
      int c0 = kb + l4, c1 = kb + 16 + l4;
      madd0 = (c0 < mask_len) ? ((amask[c0] > 0.0f) ? 1.0f : 0.0f) : 1.0f;
      madd1 = (c1 < mask_len) ? ((amask[c1] > 0.0f) ? 1.0f : 0.0f) : 1.0f;
    }
    // ---- online softmax (rows 0-7 live in lanes 0-15, rows 8-15 in 16-31)
    float alpha[8];
#pragma unroll
    for (int r = 0; r < 8; ++r) {
      float v0 = s0[r] * qk_scale + madd0;
      float v1 = s1[r] * qk_scale + madd1;
      float rmax = fmaxf(half_group_max(v0), half_group_max(v1));
      float nm = fmaxf(mrow[r], rmax);
      alpha[r] = __expf(mrow[r] - nm);
      mrow[r] = nm;
      float p0 = __expf(v0 - nm);
      float p1 = __expf(v1 - nm);
      s0[r] = p0;
      s1[r] = p1;
      lrow[r] = lrow[r] * alpha[r] + half_group_sum(p0) + half_group_sum(p1);
    }
#pragma unroll
    for (int n = 0; n < 8; ++n) {
#pragma unroll
      for (int r = 0; r < 8; ++r) acc[n][r] *= alpha[r];
    }
    // ---- issue V fragment loads before the P LDS round-trip
    v16bf bv[8];
#pragma unroll
    for (int n = 0; n < 8; ++n)
      bv[n] = *(const v16bf*)(Vh + (size_t)(n * 16 + l4) * cap_k + kb + (hf ? 16 : 0));
    SCHED_FENCE();
    // ---- P: C-layout -> LDS -> A-fragment layout (bf16); same-wave DS ops
    //      are in-order, so no barrier is required.
#pragma unroll
    for (int r = 0; r < 8; ++r) {
      int row = r + hf * 8;
      Pl[row * 32 + l4] = (__bf16)s0[r];
      Pl[row * 32 + 16 + l4] = (__bf16)s1[r];
    }
    v16bf ap = load_afrag16(&Pl[l4 * 32 + (hf ? 8 : 0)]);
    // ---- O += P @ V
#pragma unroll
    for (int n = 0; n < 8; ++n) acc[n] = wmma_bf16f32(ap, bv[n], acc[n]);
  }

  // ---- epilogue: divide by row sums, scatter to merged [row][h*128+d] bf16
#pragma unroll
  for (int r = 0; r < 8; ++r) {
    int row = q0 + r + hf * 8;
    float inv = 1.0f / lrow[r];
    __bf16* d;
    int rr;
    if (row < split) {
      d = dst_lo;
      rr = row;
    } else {
      d = dst_hi;
      rr = row - split;
    }
#pragma unroll
    for (int n = 0; n < 8; ++n) {
      int col = h * HDIM + n * 16 + l4;
      d[(size_t)rr * DMODEL + col] = (__bf16)(acc[n][r] * inv);
    }
  }
}

// ---------------------------------------------------------------------------
// host orchestration
// ---------------------------------------------------------------------------
extern "C" void kernel_launch(void* const* d_in, const int* in_sizes, int n_in,
                              void* d_out, int out_size, void* d_ws,
                              size_t ws_size, hipStream_t stream) {
  (void)in_sizes; (void)n_in; (void)out_size; (void)ws_size;

  const float* hidden  = (const float*)d_in[0];
  const float* encoder = (const float*)d_in[1];
  const float* amask   = (const float*)d_in[2];
  const float* cosT    = (const float*)d_in[3];
  const float* sinT    = (const float*)d_in[4];
  const float* lscale  = (const float*)d_in[5];
  const float* Wq  = (const float*)d_in[6];   const float* bq   = (const float*)d_in[7];
  const float* Wk  = (const float*)d_in[8];   const float* bk   = (const float*)d_in[9];
  const float* Wv  = (const float*)d_in[10];  const float* bv   = (const float*)d_in[11];
  const float* Waq = (const float*)d_in[12];  const float* baq  = (const float*)d_in[13];
  const float* Wak = (const float*)d_in[14];  const float* bak  = (const float*)d_in[15];
  const float* Wav = (const float*)d_in[16];  const float* bav  = (const float*)d_in[17];
  const float* Wo  = (const float*)d_in[18];  const float* bo   = (const float*)d_in[19];
  const float* Wad = (const float*)d_in[20];  const float* badd = (const float*)d_in[21];
  const float* qdn = (const float*)d_in[22];  const float* qup  = (const float*)d_in[23];
  const float* kdn = (const float*)d_in[24];  const float* kup  = (const float*)d_in[25];
  const float* vdn = (const float*)d_in[26];  const float* vup  = (const float*)d_in[27];
  const float* nq  = (const float*)d_in[28];  const float* nk   = (const float*)d_in[29];
  const float* naq = (const float*)d_in[30];  const float* nak  = (const float*)d_in[31];

  unsigned char* wp = (unsigned char*)d_ws;
  auto carve = [&](size_t bytes) -> void* {
    void* r = (void*)wp;
    wp += (bytes + 255) & ~(size_t)255;
    return r;
  };
  const size_t DD = (size_t)DMODEL * DMODEL;
  const size_t LORA = (size_t)RANK_C * DMODEL;

  __bf16* hb   = (__bf16*)carve((size_t)SQ_C * DMODEL * 2);   // img rows 0..2047, ref 2048..2559
  __bf16* rb   = hb + (size_t)S_IMG_C * DMODEL;
  __bf16* eb   = (__bf16*)carve((size_t)S_ENC_C * DMODEL * 2);
  __bf16* Wqb  = (__bf16*)carve(DD * 2);
  __bf16* Wkb  = (__bf16*)carve(DD * 2);
  __bf16* Wvb  = (__bf16*)carve(DD * 2);
  __bf16* Waqb = (__bf16*)carve(DD * 2);
  __bf16* Wakb = (__bf16*)carve(DD * 2);
  __bf16* Wavb = (__bf16*)carve(DD * 2);
  __bf16* Wob  = (__bf16*)carve(DD * 2);
  __bf16* Wadb = (__bf16*)carve(DD * 2);
  __bf16* qdnb = (__bf16*)carve(LORA * 2);
  __bf16* qupb = (__bf16*)carve(LORA * 2);
  __bf16* kdnb = (__bf16*)carve(LORA * 2);
  __bf16* kupb = (__bf16*)carve(LORA * 2);
  __bf16* vdnb = (__bf16*)carve(LORA * 2);
  __bf16* vupb = (__bf16*)carve(LORA * 2);
  __bf16* tb   = (__bf16*)carve((size_t)S_REF_C * RANK_C * 2);
  __bf16* Qm   = (__bf16*)carve((size_t)HEADS * SQ_C * HDIM * 2);
  __bf16* Km   = (__bf16*)carve((size_t)HEADS * SK_C * HDIM * 2);
  __bf16* Vmt  = (__bf16*)carve((size_t)HEADS * HDIM * SK_C * 2);
  __bf16* Rq   = (__bf16*)carve((size_t)HEADS * S_REF_C * HDIM * 2);
  __bf16* Rk   = (__bf16*)carve((size_t)HEADS * S_REF_C * HDIM * 2);
  __bf16* Rvt  = (__bf16*)carve((size_t)HEADS * HDIM * S_REF_C * 2);
  __bf16* Ximg = (__bf16*)carve((size_t)SQ_C * DMODEL * 2);
  __bf16* Xenc = (__bf16*)carve((size_t)S_ENC_C * DMODEL * 2);
  float*  F1   = (float*)carve((size_t)S_IMG_C * DMODEL * 4);
  float*  F2   = (float*)carve((size_t)S_REF_C * RANK_C * 4);

  auto cvt = [&](const float* s, __bf16* dst, size_t n) {
    cvt_f32_to_bf16<<<dim3(2048), dim3(256), 0, stream>>>(s, dst, n);
  };
  auto gemm = [&](const __bf16* A, const __bf16* B, const float* bias, float* C,
                  int M, int N, int Kd, int accum) {
    gemm_bf16_wmma<<<dim3(N / 128, M / 128), dim3(256), 0, stream>>>(
        A, B, bias, C, M, N, Kd, accum);
  };
  auto post = [&](const float* X, int S, const float* nrm, const float* c,
                  const float* s, int pos0, __bf16* d1, int cap1, int r01, int t1,
                  __bf16* d2, int cap2, int r02, int t2, const float* sc2) {
    qkv_post<<<dim3(S, HEADS), dim3(HDIM), 0, stream>>>(
        X, nrm, c, s, pos0, d1, cap1, r01, t1, d2, cap2, r02, t2, sc2);
  };

  // fp32 -> bf16 staging
  cvt(hidden, hb, (size_t)SQ_C * DMODEL);
  cvt(encoder, eb, (size_t)S_ENC_C * DMODEL);
  cvt(Wq, Wqb, DD);  cvt(Wk, Wkb, DD);  cvt(Wv, Wvb, DD);
  cvt(Waq, Waqb, DD); cvt(Wak, Wakb, DD); cvt(Wav, Wavb, DD);
  cvt(Wo, Wob, DD);  cvt(Wad, Wadb, DD);
  cvt(qdn, qdnb, LORA); cvt(qup, qupb, LORA);
  cvt(kdn, kdnb, LORA); cvt(kup, kupb, LORA);
  cvt(vdn, vdnb, LORA); cvt(vup, vupb, LORA);

  // image-stream q/k/v (rope positions 512..2559)
  gemm(hb, Wqb, bq, F1, S_IMG_C, DMODEL, DMODEL, 0);
  post(F1, S_IMG_C, nq, cosT, sinT, S_ENC_C, Qm, SQ_C, S_ENC_C, 0,
       nullptr, 0, 0, 0, nullptr);
  gemm(hb, Wkb, bk, F1, S_IMG_C, DMODEL, DMODEL, 0);
  post(F1, S_IMG_C, nk, cosT, sinT, S_ENC_C, Km, SK_C, S_ENC_C, 0,
       nullptr, 0, 0, 0, nullptr);
  gemm(hb, Wvb, bv, F1, S_IMG_C, DMODEL, DMODEL, 0);
  post(F1, S_IMG_C, nullptr, nullptr, nullptr, 0, Vmt, SK_C, S_ENC_C, 1,
       nullptr, 0, 0, 0, nullptr);

  // encoder q/k/v (rope positions 0..511)
  gemm(eb, Waqb, baq, F1, S_ENC_C, DMODEL, DMODEL, 0);
  post(F1, S_ENC_C, naq, cosT, sinT, 0, Qm, SQ_C, 0, 0, nullptr, 0, 0, 0, nullptr);
  gemm(eb, Wakb, bak, F1, S_ENC_C, DMODEL, DMODEL, 0);
  post(F1, S_ENC_C, nak, cosT, sinT, 0, Km, SK_C, 0, 0, nullptr, 0, 0, 0, nullptr);
  gemm(eb, Wavb, bav, F1, S_ENC_C, DMODEL, DMODEL, 0);
  post(F1, S_ENC_C, nullptr, nullptr, nullptr, 0, Vmt, SK_C, 0, 1,
       nullptr, 0, 0, 0, nullptr);

  // reference-stream q/k/v with LoRA (rope positions 2560..3071)
  gemm(rb, Wqb, bq, F1, S_REF_C, DMODEL, DMODEL, 0);
  gemm(rb, qdnb, nullptr, F2, S_REF_C, RANK_C, DMODEL, 0);
  cvt(F2, tb, (size_t)S_REF_C * RANK_C);
  gemm(tb, qupb, nullptr, F1, S_REF_C, DMODEL, RANK_C, 1);
  post(F1, S_REF_C, nq, cosT, sinT, SQ_C, Rq, S_REF_C, 0, 0,
       nullptr, 0, 0, 0, nullptr);

  gemm(rb, Wkb, bk, F1, S_REF_C, DMODEL, DMODEL, 0);
  gemm(rb, kdnb, nullptr, F2, S_REF_C, RANK_C, DMODEL, 0);
  cvt(F2, tb, (size_t)S_REF_C * RANK_C);
  gemm(tb, kupb, nullptr, F1, S_REF_C, DMODEL, RANK_C, 1);
  post(F1, S_REF_C, nk, cosT, sinT, SQ_C, Rk, S_REF_C, 0, 0,
       Km, SK_C, SQ_C, 0, lscale);  // unscaled -> ref attn; ls-scaled -> cat_k tail

  gemm(rb, Wvb, bv, F1, S_REF_C, DMODEL, DMODEL, 0);
  gemm(rb, vdnb, nullptr, F2, S_REF_C, RANK_C, DMODEL, 0);
  cvt(F2, tb, (size_t)S_REF_C * RANK_C);
  gemm(tb, vupb, nullptr, F1, S_REF_C, DMODEL, RANK_C, 1);
  post(F1, S_REF_C, nullptr, nullptr, nullptr, 0, Rvt, S_REF_C, 0, 1,
       Vmt, SK_C, SQ_C, 1, lscale);

  // attention
  flash_attn_wmma<<<dim3(SQ_C / 16, HEADS), dim3(32), 0, stream>>>(
      Qm, Km, Vmt, SQ_C, SK_C, SK_C, amask, SQ_C, Xenc, Ximg, S_ENC_C);
  flash_attn_wmma<<<dim3(S_REF_C / 16, HEADS), dim3(32), 0, stream>>>(
      Rq, Rk, Rvt, S_REF_C, S_REF_C, S_REF_C, nullptr, 0, nullptr,
      Ximg + (size_t)S_IMG_C * DMODEL, 0);

  // output projections -> d_out (hidden then enc, fp32)
  float* out_hidden = (float*)d_out;
  float* out_enc = out_hidden + (size_t)SQ_C * DMODEL;
  gemm(Ximg, Wob, bo, out_hidden, SQ_C, DMODEL, DMODEL, 0);
  gemm(Xenc, Wadb, badd, out_enc, S_ENC_C, DMODEL, DMODEL, 0);
}
